// HierarchicalGNN_77275051590045
// MI455X (gfx1250) — matrix operand order repaired and checked by reading, not compiled, not run on hardware
//
#include <hip/hip_runtime.h>
#include <math.h>

#define NN 100000
#define EE 1600000
#define GG 8
#define HH 128
#define CNND 768
#define MORPHD 6
#define DD (CNND + MORPHD)   // 774
#define NH (NN * HH)         // 12,800,000

typedef __bf16 bf16;
typedef __attribute__((ext_vector_type(16))) __bf16 v16bf;
typedef __attribute__((ext_vector_type(8)))  __bf16 v8bf;
typedef __attribute__((ext_vector_type(8)))  float  v8f;

// ---------------------------------------------------------------- utilities

__global__ void zero_kernel(float* __restrict__ p, int n) {
  int i = blockIdx.x * blockDim.x + threadIdx.x;
  if (i < n) p[i] = 0.0f;
}

__global__ void zero4_kernel(float4* __restrict__ p, int n4) {
  int i = blockIdx.x * blockDim.x + threadIdx.x;
  if (i < n4) p[i] = make_float4(0.f, 0.f, 0.f, 0.f);
}

// pack W (K x 128, row-major fp32) -> Wp (128 x K, col-major bf16)
__global__ void pack_w_kernel(const float* __restrict__ W, bf16* __restrict__ Wp, int K) {
  int idx = blockIdx.x * blockDim.x + threadIdx.x;   // col*K + k
  if (idx >= HH * K) return;
  int col = idx / K, k = idx - col * K;
  Wp[idx] = (bf16)W[(size_t)k * HH + col];
}

// deg[dst] += 1 over edges
__global__ void edge_deg_kernel(const int* __restrict__ dst, float* __restrict__ deg) {
  int e = blockIdx.x * blockDim.x + threadIdx.x;
  if (e < EE) atomicAdd(&deg[dst[e]], 1.0f);
}

__global__ void dinv_kernel(const float* __restrict__ deg, float* __restrict__ dinv) {
  int i = blockIdx.x * blockDim.x + threadIdx.x;
  if (i < NN) dinv[i] = rsqrtf(deg[i] + 1.0f);
}

// ------------------------------------------------------- fragment helpers

__device__ __forceinline__ v16bf pack_a(float4 a0, float4 a1, float4 a2, float4 a3) {
  v16bf a;
  a[0]  = (bf16)a0.x; a[1]  = (bf16)a0.y; a[2]  = (bf16)a0.z; a[3]  = (bf16)a0.w;
  a[4]  = (bf16)a1.x; a[5]  = (bf16)a1.y; a[6]  = (bf16)a1.z; a[7]  = (bf16)a1.w;
  a[8]  = (bf16)a2.x; a[9]  = (bf16)a2.y; a[10] = (bf16)a2.z; a[11] = (bf16)a2.w;
  a[12] = (bf16)a3.x; a[13] = (bf16)a3.y; a[14] = (bf16)a3.z; a[15] = (bf16)a3.w;
  return a;
}

__device__ __forceinline__ v16bf load_b16(const bf16* __restrict__ p) {  // 16B-aligned
  const v8bf* q = (const v8bf*)p;
  v8bf lo = q[0], hi = q[2];
  return __builtin_shufflevector(lo, hi, 0,1,2,3,4,5,6,7,8,9,10,11,12,13,14,15);
}

// ---------------------------------------------------------------- WMMA GEMMs
// Block = 128 threads = 4 waves; 16 rows x 128 cols of Y per block.
// Wave w owns cols [32w, 32w+32) as two 16x16 tiles; K tiled by 32 (bf16, f32 acc).
// B comes from pre-packed col-major bf16: fragment = 2 contiguous 16B loads.

// Layer-1 GEMM: Y = concat(cnn, morph) @ W.
// cnn part: 24 exact 32-K WMMA tiles from packed Wp (128 x 768 bf16).
// morph part (K=6): exact fp32 FMA tail from original W.
__global__ void gemm1_kernel(const float* __restrict__ cnn,
                             const float* __restrict__ morph,
                             const bf16*  __restrict__ Wp,
                             const float* __restrict__ W,
                             float* __restrict__ Y) {
  const int lane  = threadIdx.x & 31;
  const int wave  = threadIdx.x >> 5;
  const int row0  = blockIdx.x * 16;
  const int arow  = row0 + (lane & 15);
  const int khalf = (lane >> 4) * 8;     // A: lanes 16-31 hold K+8..15 of each 16-group
  const int bk    = (lane >> 4) * 8;     // B: lanes 16-31 hold K+8..15 of each 16-group
  const int c0col = wave * 32 + (lane & 15);
  const int c1col = c0col + 16;
  const int rbase = row0 + 8 * (lane >> 4);

  v8f c0 = {}; v8f c1 = {};
  const float* arp = cnn + (size_t)arow * CNND;
  const bf16* wp0 = Wp + (size_t)c0col * CNND + bk;
  const bf16* wp1 = Wp + (size_t)c1col * CNND + bk;
  for (int kt = 0; kt < 24; ++kt) {      // 24 * 32 = 768 exact
    const int k0 = kt * 32;
    const float4* pa = (const float4*)(arp + k0 + khalf);        // 8 floats
    const float4* pb = (const float4*)(arp + k0 + 16 + khalf);   // 8 floats
    v16bf a  = pack_a(pa[0], pa[1], pb[0], pb[1]);
    v16bf b0 = load_b16(wp0 + k0);
    v16bf b1 = load_b16(wp1 + k0);
    c0 = __builtin_amdgcn_wmma_f32_16x16x32_bf16(false, a, false, b0, (short)0, c0, false, false);
    c1 = __builtin_amdgcn_wmma_f32_16x16x32_bf16(false, a, false, b1, (short)0, c1, false, false);
  }

  // exact fp32 morph tail: k = 768..773
  float wc0[MORPHD], wc1[MORPHD];
#pragma unroll
  for (int j = 0; j < MORPHD; ++j) {
    wc0[j] = W[(size_t)(CNND + j) * HH + c0col];
    wc1[j] = W[(size_t)(CNND + j) * HH + c1col];
  }
#pragma unroll
  for (int v = 0; v < 8; ++v) {
    const float* mrow = morph + (size_t)(rbase + v) * MORPHD;
    float acc0 = 0.f, acc1 = 0.f;
#pragma unroll
    for (int j = 0; j < MORPHD; ++j) {
      float m = mrow[j];
      acc0 = fmaf(m, wc0[j], acc0);
      acc1 = fmaf(m, wc1[j], acc1);
    }
    c0[v] += acc0; c1[v] += acc1;
  }

#pragma unroll
  for (int v = 0; v < 8; ++v) {
    Y[(size_t)(rbase + v) * HH + c0col] = c0[v];
    Y[(size_t)(rbase + v) * HH + c1col] = c1[v];
  }
}

// Generic 128-K GEMM: Y = X @ W, with W pre-packed (128 x 128 col-major bf16).
__global__ void gemm128_kernel(const float* __restrict__ X,
                               const bf16*  __restrict__ Wp,
                               float* __restrict__ Y) {
  const int t     = threadIdx.x;
  const int lane  = t & 31;
  const int wave  = t >> 5;
  const int row0  = blockIdx.x * 16;
  const int arow  = row0 + (lane & 15);
  const int khalf = (lane >> 4) * 8;
  const int bk    = (lane >> 4) * 8;
  const int c0col = wave * 32 + (lane & 15);
  const int c1col = c0col + 16;
  const int rbase = row0 + 8 * (lane >> 4);

  v8f c0 = {}; v8f c1 = {};
  const float* arp = X + (size_t)arow * HH;
  const bf16* wp0 = Wp + (size_t)c0col * HH + bk;
  const bf16* wp1 = Wp + (size_t)c1col * HH + bk;
#pragma unroll
  for (int kt = 0; kt < 4; ++kt) {       // 4 * 32 = 128
    const int k0 = kt * 32;
    const float4* pa = (const float4*)(arp + k0 + khalf);
    const float4* pb = (const float4*)(arp + k0 + 16 + khalf);
    v16bf a  = pack_a(pa[0], pa[1], pb[0], pb[1]);
    v16bf b0 = load_b16(wp0 + k0);
    v16bf b1 = load_b16(wp1 + k0);
    c0 = __builtin_amdgcn_wmma_f32_16x16x32_bf16(false, a, false, b0, (short)0, c0, false, false);
    c1 = __builtin_amdgcn_wmma_f32_16x16x32_bf16(false, a, false, b1, (short)0, c1, false, false);
  }
#pragma unroll
  for (int v = 0; v < 8; ++v) {
    Y[(size_t)(rbase + v) * HH + c0col] = c0[v];
    Y[(size_t)(rbase + v) * HH + c1col] = c1[v];
  }
}

// ---------------------------------------------------------------- edges

// one wave32 per edge; each lane moves float4 (128 features total)
__global__ void edge_msg_kernel(const int* __restrict__ src, const int* __restrict__ dst,
                                const float* __restrict__ dinv,
                                const float* __restrict__ h,
                                float* __restrict__ agg) {
  const int e = (blockIdx.x * blockDim.x + threadIdx.x) >> 5;
  if (e >= EE) return;
  const int lane = threadIdx.x & 31;
  const int s = src[e], d = dst[e];
  const float nrm = dinv[s] * dinv[d];
  const float4 v = ((const float4*)(h + (size_t)s * HH))[lane];
  float* ap = agg + (size_t)d * HH + lane * 4;
  atomicAdd(ap + 0, v.x * nrm);
  atomicAdd(ap + 1, v.y * nrm);
  atomicAdd(ap + 2, v.z * nrm);
  atomicAdd(ap + 3, v.w * nrm);
}

// cur = relu(bn(agg + h*dinv^2 + b)); xout (+)= cur   (float4 per thread)
__global__ void post_kernel(const float4* __restrict__ agg, const float4* __restrict__ h,
                            const float* __restrict__ dinv, const float4* __restrict__ bias,
                            const float4* __restrict__ gamma, const float4* __restrict__ beta,
                            float4* __restrict__ cur, float4* __restrict__ xout, int accum) {
  int idx = blockIdx.x * blockDim.x + threadIdx.x;   // over NH/4
  if (idx >= NH / 4) return;
  const int i  = idx >> 5;          // node (32 float4 per row)
  const int f4 = idx & 31;          // float4 index within row
  const float di = dinv[i];
  const float d2 = di * di;
  const float s  = rsqrtf(1.0f + 1e-5f);
  float4 av = agg[idx], hv = h[idx], bv = bias[f4], gv = gamma[f4], ev = beta[f4];
  float4 v;
  v.x = fmaxf(fmaf(av.x + hv.x * d2 + bv.x, gv.x * s, ev.x), 0.f);
  v.y = fmaxf(fmaf(av.y + hv.y * d2 + bv.y, gv.y * s, ev.y), 0.f);
  v.z = fmaxf(fmaf(av.z + hv.z * d2 + bv.z, gv.z * s, ev.z), 0.f);
  v.w = fmaxf(fmaf(av.w + hv.w * d2 + bv.w, gv.w * s, ev.w), 0.f);
  cur[idx] = v;
  if (accum) {
    float4 xo = xout[idx];
    xo.x += v.x; xo.y += v.y; xo.z += v.z; xo.w += v.w;
    xout[idx] = xo;
  } else {
    xout[idx] = v;
  }
}

// ---------------------------------------------------------------- attention

__global__ void tanh_bias_kernel(float4* __restrict__ t, const float4* __restrict__ b) {
  int idx = blockIdx.x * blockDim.x + threadIdx.x;   // over NH/4
  if (idx >= NH / 4) return;
  float4 tv = t[idx], bv = b[idx & 31];
  tv.x = tanhf(tv.x + bv.x); tv.y = tanhf(tv.y + bv.y);
  tv.z = tanhf(tv.z + bv.z); tv.w = tanhf(tv.w + bv.w);
  t[idx] = tv;
}

// gate[i] = dot(t[i,:], w2) + b2  (one wave per node)
__global__ void gate_kernel(const float* __restrict__ t, const float* __restrict__ w2,
                            const float* __restrict__ b2, float* __restrict__ gate) {
  const int i = (blockIdx.x * blockDim.x + threadIdx.x) >> 5;
  if (i >= NN) return;
  const int lane = threadIdx.x & 31;
  const float4 tv = ((const float4*)(t + (size_t)i * HH))[lane];
  const float4 wv = ((const float4*)w2)[lane];
  float s = tv.x * wv.x + tv.y * wv.y + tv.z * wv.z + tv.w * wv.w;
#pragma unroll
  for (int off = 16; off; off >>= 1) s += __shfl_xor(s, off, 32);
  if (lane == 0) gate[i] = s + b2[0];
}

__device__ __forceinline__ unsigned fenc(float f) {
  unsigned u = __float_as_uint(f);
  return (u >> 31) ? ~u : (u | 0x80000000u);
}
__device__ __forceinline__ float fdec(unsigned k) {
  return (k >> 31) ? __uint_as_float(k ^ 0x80000000u) : __uint_as_float(~k);
}

__global__ void gmax_kernel(const int* __restrict__ batch, const float* __restrict__ gate,
                            unsigned* __restrict__ gmaxU) {
  int i = blockIdx.x * blockDim.x + threadIdx.x;
  if (i < NN) atomicMax(&gmaxU[batch[i]], fenc(gate[i]));
}

// gate[i] <- exp(gate[i]-gmax[b]);  denom[b] += a  (LDS two-stage)
__global__ void softmax_a_kernel(const int* __restrict__ batch,
                                 const unsigned* __restrict__ gmaxU,
                                 float* __restrict__ gate, float* __restrict__ denom) {
  __shared__ float dl[GG];
  if (threadIdx.x < GG) dl[threadIdx.x] = 0.0f;
  __syncthreads();
  int i = blockIdx.x * blockDim.x + threadIdx.x;
  if (i < NN) {
    int g = batch[i];
    float a = expf(gate[i] - fdec(gmaxU[g]));
    gate[i] = a;
    atomicAdd(&dl[g], a);
  }
  __syncthreads();
  if (threadIdx.x < GG) atomicAdd(&denom[threadIdx.x], dl[threadIdx.x]);
}

#define POOL_NODES 256
// pooled[g,f] = sum_i (a_i/denom_g) * xout[i,f]  — LDS partials per block
__global__ void pooled_kernel(const int* __restrict__ batch, const float* __restrict__ gate,
                              const float* __restrict__ denom, const float* __restrict__ xout,
                              float* __restrict__ pooled) {
  __shared__ float acc[GG * HH];
  const int f = threadIdx.x;   // 128 threads
#pragma unroll
  for (int g = 0; g < GG; ++g) acc[g * HH + f] = 0.0f;
  int n0 = blockIdx.x * POOL_NODES;
  int n1 = n0 + POOL_NODES; if (n1 > NN) n1 = NN;
  for (int n = n0; n < n1; ++n) {
    int g = batch[n];
    float alpha = gate[n] / denom[g];
    acc[g * HH + f] += alpha * xout[(size_t)n * HH + f];
  }
#pragma unroll
  for (int g = 0; g < GG; ++g) atomicAdd(&pooled[g * HH + f], acc[g * HH + f]);
}

// ---------------------------------------------------------------- classifier

__global__ void cls_kernel(const float* __restrict__ pooled,
                           const float* __restrict__ w1, const float* __restrict__ b1,
                           const float* __restrict__ g1, const float* __restrict__ be1,
                           const float* __restrict__ w2, const float* __restrict__ b2,
                           const float* __restrict__ g2, const float* __restrict__ be2,
                           const float* __restrict__ w3, const float* __restrict__ b3,
                           float* __restrict__ out) {
  __shared__ float h1[GG * 256];
  __shared__ float h2[GG * HH];
  const float s = rsqrtf(1.0f + 1e-5f);
  const int j = threadIdx.x;           // 256 threads
  for (int g = 0; g < GG; ++g) {
    float acc = 0.0f;
    for (int k = 0; k < HH; ++k) acc += pooled[g * HH + k] * w1[k * 256 + j];
    acc = (acc + b1[j]) * (g1[j] * s) + be1[j];
    h1[g * 256 + j] = fmaxf(acc, 0.0f);
  }
  __syncthreads();
  if (j < HH) {
    for (int g = 0; g < GG; ++g) {
      float acc = 0.0f;
      for (int k = 0; k < 256; ++k) acc += h1[g * 256 + k] * w2[k * HH + j];
      acc = (acc + b2[j]) * (g2[j] * s) + be2[j];
      h2[g * HH + j] = fmaxf(acc, 0.0f);
    }
  }
  __syncthreads();
  if (j < GG * 5) {
    int g = j / 5, c = j % 5;
    float acc = 0.0f;
    for (int k = 0; k < HH; ++k) acc += h2[g * HH + k] * w3[k * 5 + c];
    out[g * 5 + c] = acc + b3[c];
  }
}

// ---------------------------------------------------------------- launch

extern "C" void kernel_launch(void* const* d_in, const int* in_sizes, int n_in,
                              void* d_out, int out_size, void* d_ws, size_t ws_size,
                              hipStream_t stream) {
  const float* cnn   = (const float*)d_in[0];
  const float* morph = (const float*)d_in[1];
  const int*   ei    = (const int*)d_in[2];
  const int*   batch = (const int*)d_in[3];
  const float* g1w = (const float*)d_in[4];   const float* g1b = (const float*)d_in[5];
  const float* g2w = (const float*)d_in[6];   const float* g2b = (const float*)d_in[7];
  const float* g3w = (const float*)d_in[8];   const float* g3b = (const float*)d_in[9];
  const float* bn1g = (const float*)d_in[10]; const float* bn1b = (const float*)d_in[11];
  const float* bn2g = (const float*)d_in[12]; const float* bn2b = (const float*)d_in[13];
  const float* bn3g = (const float*)d_in[14]; const float* bn3b = (const float*)d_in[15];
  const float* aw1 = (const float*)d_in[16];  const float* ab1 = (const float*)d_in[17];
  const float* aw2 = (const float*)d_in[18];  const float* ab2 = (const float*)d_in[19];
  const float* cw1 = (const float*)d_in[20];  const float* cb1 = (const float*)d_in[21];
  const float* cbn1g = (const float*)d_in[22];const float* cbn1b = (const float*)d_in[23];
  const float* cw2 = (const float*)d_in[24];  const float* cb2 = (const float*)d_in[25];
  const float* cbn2g = (const float*)d_in[26];const float* cbn2b = (const float*)d_in[27];
  const float* cw3 = (const float*)d_in[28];  const float* cb3 = (const float*)d_in[29];

  const int* src = ei;
  const int* dst = ei + EE;

  float* ws    = (float*)d_ws;
  float* deg   = ws;                  // N   (reused later as gate / a)
  float* dinv  = deg  + NN;           // N
  float* h     = dinv + NN;           // N*H (also reused as tanh buffer t)
  float* agg   = h    + NH;           // N*H
  float* cur   = agg  + NH;           // N*H
  float* xout  = cur  + NH;           // N*H
  float* small = xout + NH;           // gmaxU(8) | denom(8) | pooled(1024)
  unsigned* gmaxU = (unsigned*)small;
  float* denom  = small + GG;
  float* pooled = denom + GG;
  float* gate   = deg;
  // packed bf16 weights (col-major); region is 16B-aligned
  bf16* wp1 = (bf16*)(small + GG + GG + GG * HH);  // 128*768
  bf16* wp2 = wp1 + HH * CNND;                     // 128*128
  bf16* wp3 = wp2 + HH * HH;
  bf16* wpa = wp3 + HH * HH;

  const int B = 256;
  const int nh4Blocks = (NH / 4 + B - 1) / B;     // 12500
  const int nBlocks   = (NN + B - 1) / B;         // 391
  const int eBlocks   = (EE + B - 1) / B;         // 6250
  const int ewBlocks  = (EE * 32) / B;            // 200000 (one wave per edge)
  const int gwBlocks  = (NN * 32 + B - 1) / B;    // 12500
  const int gemmBlocks = NN / 16;                 // 6250
  const int poolBlocks = (NN + POOL_NODES - 1) / POOL_NODES;

  // pack weights to bf16 col-major (one shot, ~150K elements total)
  pack_w_kernel<<<(HH * CNND + B - 1) / B, B, 0, stream>>>(g1w, wp1, CNND);
  pack_w_kernel<<<(HH * HH + B - 1) / B, B, 0, stream>>>(g2w, wp2, HH);
  pack_w_kernel<<<(HH * HH + B - 1) / B, B, 0, stream>>>(g3w, wp3, HH);
  pack_w_kernel<<<(HH * HH + B - 1) / B, B, 0, stream>>>(aw1, wpa, HH);

  // degree + dinv
  zero_kernel<<<nBlocks, B, 0, stream>>>(deg, NN);
  edge_deg_kernel<<<eBlocks, B, 0, stream>>>(dst, deg);
  dinv_kernel<<<nBlocks, B, 0, stream>>>(deg, dinv);

  // layer 1 GEMM: h = concat(cnn,morph) @ W1
  gemm1_kernel<<<gemmBlocks, 128, 0, stream>>>(cnn, morph, wp1, g1w, h);

  const float* lb[3]  = {g1b, g2b, g3b};
  const float* lg[3]  = {bn1g, bn2g, bn3g};
  const float* lbe[3] = {bn1b, bn2b, bn3b};
  const bf16*  lw[2]  = {wp2, wp3};
  for (int l = 0; l < 3; ++l) {
    zero4_kernel<<<nh4Blocks, B, 0, stream>>>((float4*)agg, NH / 4);
    edge_msg_kernel<<<ewBlocks, B, 0, stream>>>(src, dst, dinv, h, agg);
    post_kernel<<<nh4Blocks, B, 0, stream>>>((const float4*)agg, (const float4*)h, dinv,
                                             (const float4*)lb[l], (const float4*)lg[l],
                                             (const float4*)lbe[l],
                                             (float4*)cur, (float4*)xout, l > 0);
    if (l < 2)
      gemm128_kernel<<<gemmBlocks, 128, 0, stream>>>(cur, lw[l], h);
  }

  // attention: t = tanh(xout @ aw1 + ab1); gate = t @ aw2 + ab2
  gemm128_kernel<<<gemmBlocks, 128, 0, stream>>>(xout, wpa, h);
  tanh_bias_kernel<<<nh4Blocks, B, 0, stream>>>((float4*)h, (const float4*)ab1);
  gate_kernel<<<gwBlocks, B, 0, stream>>>(h, aw2, ab2, gate);

  // segment softmax + weighted pooling
  zero_kernel<<<1, GG + GG + GG * HH, 0, stream>>>(small, GG + GG + GG * HH);
  gmax_kernel<<<nBlocks, B, 0, stream>>>(batch, gate, gmaxU);
  softmax_a_kernel<<<nBlocks, B, 0, stream>>>(batch, gmaxU, gate, denom);
  pooled_kernel<<<poolBlocks, HH, 0, stream>>>(batch, gate, denom, xout, pooled);

  // classifier head
  cls_kernel<<<1, 256, 0, stream>>>(pooled, cw1, cb1, cbn1g, cbn1b,
                                    cw2, cb2, cbn2g, cbn2b, cw3, cb3,
                                    (float*)d_out);
}
